// MessagePassing_30477087933114
// MI455X (gfx1250) — matrix operand order repaired and checked by reading, not compiled, not run on hardware
//
#include <hip/hip_runtime.h>
#include <hip/hip_bf16.h>

typedef __attribute__((ext_vector_type(16))) _Float16 v16h;
typedef __attribute__((ext_vector_type(8)))  float    v8f;

// w = exp(-1)
#define KNN_W 0.36787944117144233f

// ---------------------------------------------------------------------------
// out[N,64] = relu( A[N, KCHUNKS*32] (row stride lda) @ W[KCHUNKS*32, 64] + b )
// One wave computes a 16-row x 64-col slab via v_wmma_f32_16x16x32_f16.
// W is converted f32->f16 ONCE per block into LDS, laid out exactly as the
// wave32 B-fragments, so each WMMA's B operand is one 32B LDS read.
//
// Fragment layouts (CDNA5 ISA, wave32):
//   A 16x32 f16: lanes 0-15 -> M=lane,  K {hs*8+0..7, hs*8+16..23} (hs=lane>>4)
//   B 32x16 f16: lanes 0-15 -> N=lane,  K {hs*8+0..7, hs*8+16..23}, same as A
//   C/D 16x16 f32: VGPR r -> M=r (lanes 0-15), M=r+8 (lanes 16-31); N=lane&15
// ---------------------------------------------------------------------------
template <int KCHUNKS>
__global__ __launch_bounds__(256) void gemm_relu_wmma(
    const float* __restrict__ A, long lda,
    const float* __restrict__ W, const float* __restrict__ bias,
    float* __restrict__ out, int N)
{
  // ---- cooperative W -> f16 fragment staging in LDS ----
  // fragment id = ck*4 + tile; each fragment = 32 lanes x 16 halves (512 h)
  __shared__ _Float16 Wh[KCHUNKS * 4 * 32 * 16];
#pragma unroll
  for (int i = threadIdx.x; i < KCHUNKS * 4 * 512; i += 256) {
    const int e    = i & 15;
    const int ln   = (i >> 4) & 31;
    const int frag = i >> 9;
    const int ck   = frag >> 2;
    const int tile = frag & 3;
    const int col  = tile * 16 + (ln & 15);
    const int kb   = ck * 32 + (ln >> 4) * 8 + (e >> 3) * 16 + (e & 7);
    Wh[i] = (_Float16)W[kb * 64 + col];
  }
  __syncthreads();

  const int lane = threadIdx.x & 31;
  const int wave = threadIdx.x >> 5;
  const int m0   = (blockIdx.x * 8 + wave) * 16;
  if (m0 >= N) return;                 // wave-uniform: EXEC all-ones for WMMA

  const int halfSel = lane >> 4;
  const int l15     = lane & 15;

  // ---- A fragments (held in registers across all 4 column tiles) ----
  int rowA = m0 + l15;
  if (rowA >= N) rowA = N - 1;
  v16h afrag[KCHUNKS];
#pragma unroll
  for (int c = 0; c < KCHUNKS; ++c) {
    const float* ap = A + (long)rowA * lda + c * 32 + halfSel * 8;
    v16h a;
#pragma unroll
    for (int e = 0; e < 8; ++e) {
      a[e]     = (_Float16)ap[e];
      a[8 + e] = (_Float16)ap[16 + e];
    }
    afrag[c] = a;
  }

  const bool fullTile = (m0 + 16 <= N);

#pragma unroll
  for (int t = 0; t < 4; ++t) {
    const int n0  = t * 16;
    const int col = n0 + l15;
    const float bv = bias[col];
    v8f c;
#pragma unroll
    for (int r = 0; r < 8; ++r) c[r] = bv;

#pragma unroll
    for (int ck = 0; ck < KCHUNKS; ++ck) {
      const v16h b = *(const v16h*)&Wh[((ck << 2) + t) * 512 + lane * 16];
      c = __builtin_amdgcn_wmma_f32_16x16x32_f16(
              false, afrag[ck], false, b, (short)0, c, false, false);
    }

    float* op = out + (long)(m0 + halfSel * 8) * 64 + col;
    if (fullTile) {
#pragma unroll
      for (int r = 0; r < 8; ++r) {
        const float v = c[r];
        op[(long)r * 64] = v > 0.f ? v : 0.f;
      }
    } else {
#pragma unroll
      for (int r = 0; r < 8; ++r) {
        if (m0 + halfSel * 8 + r < N) {
          const float v = c[r];
          op[(long)r * 64] = v > 0.f ? v : 0.f;
        }
      }
    }
  }
}

// ---------------------------------------------------------------------------
// KNN mean/max accumulate: 64 threads per row, feature-per-thread.
// Gathered rows are 256B contiguous -> coalesced; f (51 MB) stays in 192MB L2.
// NT=true marks final-output stores non-temporal so the 256MB output stream
// does not evict the L2-resident gather working set.
// ---------------------------------------------------------------------------
template <bool NT>
__global__ __launch_bounds__(256) void knn_acc(
    const float* __restrict__ f, const long long* __restrict__ idx,
    float* __restrict__ out, int colOff, int N)
{
  __shared__ int sidx[4][32];
  const int t   = threadIdx.x & 63;
  const int r   = threadIdx.x >> 6;
  const int row = blockIdx.x * 4 + r;

  if (row < N && t < 32)
    sidx[r][t] = (int)idx[(long)row * 32 + t];
  __syncthreads();
  if (row >= N) return;

  float sum = 0.f, mx = -3.402823466e38f;
#pragma unroll 8
  for (int k = 0; k < 32; ++k) {
    const float v = f[(long)sidx[r][k] * 64 + t];
    sum += v;
    mx = fmaxf(mx, v);
  }
  const float self = f[(long)row * 64 + t];
  float* o = out + (long)row * 320 + colOff;
  const float vmean = KNN_W * (sum * (1.f / 32.f)) - self;
  const float vmax  = KNN_W * mx - self;
  if (NT) {
    __builtin_nontemporal_store(vmean, o + t);
    __builtin_nontemporal_store(vmax,  o + 64 + t);
  } else {
    o[t]      = vmean;
    o[64 + t] = vmax;
  }
}

// ---------------------------------------------------------------------------
// Copy x into output columns 256..319 (final output -> non-temporal stores)
// ---------------------------------------------------------------------------
__global__ __launch_bounds__(256) void copy_x(
    const float* __restrict__ x, float* __restrict__ out, int N)
{
  const long i = (long)blockIdx.x * blockDim.x + threadIdx.x;
  if (i >= (long)N * 64) return;
  const long row = i >> 6;
  const int  t   = (int)(i & 63);
  __builtin_nontemporal_store(x[i], out + row * 320 + 256 + t);
}

extern "C" void kernel_launch(void* const* d_in, const int* in_sizes, int n_in,
                              void* d_out, int out_size, void* d_ws, size_t ws_size,
                              hipStream_t stream)
{
  const float*     x   = (const float*)d_in[0];
  const long long* idx = (const long long*)d_in[1];
  const float*     W0  = (const float*)d_in[2];
  const float*     b0  = (const float*)d_in[3];
  const float*     W1  = (const float*)d_in[4];
  const float*     b1  = (const float*)d_in[5];
  float* out = (float*)d_out;

  const int N = in_sizes[0] / 64;
  float* fbuf = (float*)d_ws;            // [N,64] fp32 scratch (51.2 MB), reused

  const int  tiles = (N + 15) / 16;
  const dim3 gGemm((tiles + 7) / 8), bGemm(256);
  const dim3 gAcc((N + 3) / 4),      bAcc(256);
  const long total = (long)N * 64;
  const dim3 gCopy((unsigned)((total + 255) / 256)), bCopy(256);

  // Layer 0: f0 = relu(x @ W0 + b0); out0 -> cols 0..127 (re-read -> RT stores)
  gemm_relu_wmma<2><<<gGemm, bGemm, 0, stream>>>(x, 64, W0, b0, fbuf, N);
  knn_acc<false><<<gAcc, bAcc, 0, stream>>>(fbuf, idx, out, 0, N);

  // Layer 1: f1 = relu(out0 @ W1 + b1); input is d_out cols 0..127, stride 320
  gemm_relu_wmma<4><<<gGemm, bGemm, 0, stream>>>(out, 320, W1, b1, fbuf, N);
  knn_acc<true><<<gAcc, bAcc, 0, stream>>>(fbuf, idx, out, 128, N);

  // Pass-through x -> cols 256..319 (final output -> NT stores)
  copy_x<<<gCopy, bCopy, 0, stream>>>(x, out, N);
}
